// LSTMClassifier_29841432772857
// MI455X (gfx1250) — matrix operand order repaired and checked by reading, not compiled, not run on hardware
//
#include <hip/hip_runtime.h>
#include <hip/hip_bf16.h>

#define SEQ  8192
#define INF  256          // input features
#define HID  2048
#define NWG  128
#define THREADS 256
#define ROWS 64           // gate rows per WG (4 gates x 16 hidden)
#define HB   16           // hidden elements per WG
#define RS_HH 4112        // LDS row stride bytes for w_hh (2048*2 + 16 pad -> bank spread)
#define RS_IH 528         // LDS row stride bytes for w_ih (256*2 + 16 pad)
#define OFF_WHH 0
#define OFF_WIH (ROWS * RS_HH)                 // 263168
#define OFF_GACC (OFF_WIH + ROWS * RS_IH)      // 296960
#define OFF_BIAS (OFF_GACC + ROWS * 4)         // 297216
#define SMEM_BYTES (OFF_BIAS + ROWS * 4)       // 297472 (< 320 KB WGP LDS)

typedef __attribute__((ext_vector_type(16))) __bf16 v16bf;
typedef __attribute__((ext_vector_type(8)))  float  v8f;

__device__ __forceinline__ unsigned short f2bf(float f) {
  unsigned u = __float_as_uint(f);
  return (unsigned short)((u + 0x7FFFu + ((u >> 16) & 1u)) >> 16);   // RNE
}
__device__ __forceinline__ float bf2f(unsigned short s) {
  return __uint_as_float(((unsigned)s) << 16);
}
__device__ __forceinline__ float sigm(float x) {
  return 1.0f / (1.0f + __expf(-x));
}

// Device-wide split barrier: monotonic counter, agent-scope release/acquire.
__device__ __forceinline__ void grid_sync(int* cnt, int target) {
  __builtin_amdgcn_fence(__ATOMIC_RELEASE, "agent");
  __syncthreads();
  if (threadIdx.x == 0) {
    __hip_atomic_fetch_add(cnt, 1, __ATOMIC_RELAXED, __HIP_MEMORY_SCOPE_AGENT);
    while (__hip_atomic_load(cnt, __ATOMIC_RELAXED, __HIP_MEMORY_SCOPE_AGENT) < target)
      __builtin_amdgcn_s_sleep(2);
  }
  __syncthreads();
  __builtin_amdgcn_fence(__ATOMIC_ACQUIRE, "agent");
}

__global__ void lstm_init(const float* __restrict__ h0,
                          unsigned short* __restrict__ hbuf,
                          int* __restrict__ cnt) {
  int i = blockIdx.x * blockDim.x + threadIdx.x;
  if (i < HID) hbuf[i] = f2bf(h0[i]);
  if (i == 0) *cnt = 0;
}

__global__ void __launch_bounds__(THREADS)
lstm_persistent(const float* __restrict__ x,
                const float* __restrict__ c0,
                const float* __restrict__ w_ih,
                const float* __restrict__ w_hh,
                const float* __restrict__ b_ih,
                const float* __restrict__ b_hh,
                const float* __restrict__ w_lin,
                const float* __restrict__ b_lin,
                unsigned short* __restrict__ hbuf,   // 2 x HID bf16 (ping-pong)
                int* __restrict__ cnt,
                float* __restrict__ out) {
  extern __shared__ char smem[];
  const int tid   = threadIdx.x;
  const int wave  = tid >> 5;
  const int lane  = tid & 31;
  const int wg    = blockIdx.x;
  const int hbase = wg * HB;

  float* gacc = (float*)(smem + OFF_GACC);
  float* bias = (float*)(smem + OFF_BIAS);

  // ---------- one-time: stage bf16 weight slices into LDS ----------
  // local row m = gate*16 + j  <->  global gate row = gate*HID + hbase + j
  for (int q = tid; q < ROWS * (HID / 4); q += THREADS) {
    int m = q >> 9, kq = q & 511;
    int g = m >> 4, j = m & 15;
    const float4 f = *(const float4*)(w_hh + (size_t)(g * HID + hbase + j) * HID + kq * 4);
    uint2 p;
    p.x = (unsigned)f2bf(f.x) | ((unsigned)f2bf(f.y) << 16);
    p.y = (unsigned)f2bf(f.z) | ((unsigned)f2bf(f.w) << 16);
    *(uint2*)(smem + OFF_WHH + m * RS_HH + kq * 8) = p;
  }
  for (int q = tid; q < ROWS * (INF / 4); q += THREADS) {
    int m = q >> 6, kq = q & 63;
    int g = m >> 4, j = m & 15;
    const float4 f = *(const float4*)(w_ih + (size_t)(g * HID + hbase + j) * INF + kq * 4);
    uint2 p;
    p.x = (unsigned)f2bf(f.x) | ((unsigned)f2bf(f.y) << 16);
    p.y = (unsigned)f2bf(f.z) | ((unsigned)f2bf(f.w) << 16);
    *(uint2*)(smem + OFF_WIH + m * RS_IH + kq * 8) = p;
  }
  if (tid < ROWS) {
    int g = tid >> 4, j = tid & 15;
    int grow = g * HID + hbase + j;
    bias[tid] = b_ih[grow] + b_hh[grow];
    gacc[tid] = 0.0f;
  }
  float creg = 0.0f;
  if (wave == 0 && lane < HB) creg = c0[hbase + lane];
  __syncthreads();

  const int aoff = (lane < 16) ? 0 : 32;   // A-fragment byte select (K-half per lane group)
  const int boff = (lane < 16) ? 0 : 16;   // B-fragment element select (K 0..15 vs 16..31)
  const int m15  = lane & 15;

  #pragma unroll 1
  for (int t = 0; t < SEQ; ++t) {
    const unsigned short* hin = hbuf + (t & 1) * HID;
    const float* xt = x + (size_t)t * INF;
    if (wave == 1 && lane == 0 && t + 1 < SEQ)
      __builtin_prefetch(x + (size_t)(t + 1) * INF, 0, 0);

    v8f acc[4];
    #pragma unroll
    for (int g = 0; g < 4; ++g) acc[g] = (v8f){0.f,0.f,0.f,0.f,0.f,0.f,0.f,0.f};

    // -------- recurrent matvec: (ROWS x HID) @ h, wave owns k-tiles [wave*8, wave*8+8) --------
    #pragma unroll
    for (int i = 0; i < 8; ++i) {
      const int k0 = (wave * 8 + i) * 32;
      v16bf bv;                                   // h chunk, replicated over 16 columns
      *(uint4*)&bv       = *(const uint4*)(hin + k0 + boff);
      *((uint4*)&bv + 1) = *(const uint4*)(hin + k0 + boff + 8);
      #pragma unroll
      for (int g = 0; g < 4; ++g) {
        const char* ap = smem + OFF_WHH + (g * 16 + m15) * RS_HH + k0 * 2 + aoff;
        v16bf av;
        *(uint4*)&av       = *(const uint4*)(ap);       // K = kb..kb+7
        *((uint4*)&av + 1) = *(const uint4*)(ap + 16);  // K = kb+8..kb+15
        acc[g] = __builtin_amdgcn_wmma_f32_16x16x32_bf16(
            false, av, false, bv, (short)0, acc[g], false, false);
      }
    }

    // -------- fused input matvec: (ROWS x INF) @ x_t, one k-tile per wave --------
    {
      const int k0 = wave * 32;
      const float4* xp = (const float4*)(xt + k0 + boff);
      float4 f0 = xp[0], f1 = xp[1], f2 = xp[2], f3 = xp[3];
      v16bf bv;
      unsigned short* bs = (unsigned short*)&bv;
      bs[0]=f2bf(f0.x); bs[1]=f2bf(f0.y); bs[2] =f2bf(f0.z); bs[3] =f2bf(f0.w);
      bs[4]=f2bf(f1.x); bs[5]=f2bf(f1.y); bs[6] =f2bf(f1.z); bs[7] =f2bf(f1.w);
      bs[8]=f2bf(f2.x); bs[9]=f2bf(f2.y); bs[10]=f2bf(f2.z); bs[11]=f2bf(f2.w);
      bs[12]=f2bf(f3.x);bs[13]=f2bf(f3.y);bs[14]=f2bf(f3.z); bs[15]=f2bf(f3.w);
      #pragma unroll
      for (int g = 0; g < 4; ++g) {
        const char* ap = smem + OFF_WIH + (g * 16 + m15) * RS_IH + k0 * 2 + aoff;
        v16bf av;
        *(uint4*)&av       = *(const uint4*)(ap);
        *((uint4*)&av + 1) = *(const uint4*)(ap + 16);
        acc[g] = __builtin_amdgcn_wmma_f32_16x16x32_bf16(
            false, av, false, bv, (short)0, acc[g], false, false);
      }
    }

    // -------- cross-wave reduction (D columns are replicated; lanes 0/16 hold rows 0-7/8-15) --------
    if ((lane & 15) == 0) {
      const int rbase = (lane >> 4) * 8;
      #pragma unroll
      for (int g = 0; g < 4; ++g)
        #pragma unroll
        for (int v = 0; v < 8; ++v)
          atomicAdd(&gacc[g * 16 + rbase + v], acc[g][v]);
    }
    __syncthreads();

    // -------- gate nonlinearity + state update for our 16 hidden elements --------
    if (wave == 0 && lane < HB) {
      const int j = lane;
      float gi = gacc[j]      + bias[j];
      float gf = gacc[16 + j] + bias[16 + j];
      float gg = gacc[32 + j] + bias[32 + j];
      float go = gacc[48 + j] + bias[48 + j];
      creg = sigm(gf) * creg + sigm(gi) * tanhf(gg);
      float h = sigm(go) * tanhf(creg);
      hbuf[((t + 1) & 1) * HID + hbase + j] = f2bf(h);
      gacc[j] = 0.f; gacc[16 + j] = 0.f; gacc[32 + j] = 0.f; gacc[48 + j] = 0.f;
    }
    grid_sync(cnt, (t + 1) * NWG);
  }

  // -------- final linear: logits = h @ w_lin^T + b_lin (SEQ even -> final h in buffer 0) --------
  if (wg == 0 && wave == 0) {
    const unsigned short* hf = hbuf;
    #pragma unroll
    for (int cc = 0; cc < 2; ++cc) {
      float s = 0.f;
      for (int k = lane; k < HID; k += 32)
        s += bf2f(hf[k]) * w_lin[cc * HID + k];
      #pragma unroll
      for (int off = 16; off > 0; off >>= 1)
        s += __shfl_down(s, off);
      if (lane == 0) out[cc] = s + b_lin[cc];
    }
  }
}

extern "C" void kernel_launch(void* const* d_in, const int* in_sizes, int n_in,
                              void* d_out, int out_size, void* d_ws, size_t ws_size,
                              hipStream_t stream) {
  const float* x     = (const float*)d_in[0];
  const float* h0    = (const float*)d_in[1];
  const float* c0    = (const float*)d_in[2];
  const float* w_ih  = (const float*)d_in[3];
  const float* w_hh  = (const float*)d_in[4];
  const float* b_ih  = (const float*)d_in[5];
  const float* b_hh  = (const float*)d_in[6];
  const float* w_lin = (const float*)d_in[7];
  const float* b_lin = (const float*)d_in[8];
  float* out = (float*)d_out;

  unsigned short* hbuf = (unsigned short*)d_ws;                       // 2 x 2048 bf16
  int* cnt = (int*)((char*)d_ws + 2 * HID * sizeof(unsigned short));  // barrier counter

  (void)hipFuncSetAttribute((const void*)lstm_persistent,
                            hipFuncAttributeMaxDynamicSharedMemorySize, SMEM_BYTES);

  lstm_init<<<(HID + 255) / 256, 256, 0, stream>>>(h0, hbuf, cnt);
  lstm_persistent<<<NWG, THREADS, SMEM_BYTES, stream>>>(
      x, c0, w_ih, w_hh, b_ih, b_hh, w_lin, b_lin, hbuf, cnt, out);
}
